// WeightedMultiheadAttention_60438779789642
// MI455X (gfx1250) — compile-verified, matrix operand order checked by reading, not run on hardware
//
#include <hip/hip_runtime.h>
#include <hip/hip_bf16.h>
#include <math.h>

// ---------------------------------------------------------------------------
// Weighted multi-head attention for MI455X (gfx1250, wave32, WMMA).
//   B=2, S=2048, E=1024, H=16, hd=64.
//   Stage 1: convert x -> f16, W_qkv -> f16 transposed, bias = log(w+tiny)
//   Stage 2: QKV GEMM, 32x64 tile/wave, wmma f32_16x16x32_f16 (no spills:
//            single-buffered chunk loads + __launch_bounds__(256,1))
//   Stage 3: flash attention, one wave per (b,h,16-query tile), all-WMMA,
//            global_prefetch for next K/V tiles.
// ---------------------------------------------------------------------------

#define BATCH 2
#define SEQ   2048
#define EMB   1024
#define HEADS 16
#define HD    64

typedef __attribute__((ext_vector_type(16))) _Float16 v16h;
typedef __attribute__((ext_vector_type(8)))  float    v8f;

union H16 { v16h v; uint32_t u[8]; };

// Load one 16x32 f16 WMMA operand (A layout; B uses the same pattern when the
// source is stored "transposed" so that element (k, n) == src[n*ld + k]).
//   lane&15  -> row (M for A, N for B)
//   lane>>4  -> which K-half (K offsets +0/+8 and +16/+24)
//   pairs of consecutive-K f16 per VGPR -> compiler merges into b128 loads.
__device__ __forceinline__ v16h load_tile(const _Float16* __restrict__ src, int ld) {
    const int lane = threadIdx.x & 31;
    const int row  = lane & 15;
    const int kh   = (lane >> 4) * 8;
    const _Float16* p = src + (size_t)row * ld;
    H16 t;
#pragma unroll
    for (int i = 0; i < 4; ++i)
        t.u[i]     = *(const uint32_t*)(p + kh + 2 * i);
#pragma unroll
    for (int i = 0; i < 4; ++i)
        t.u[4 + i] = *(const uint32_t*)(p + 16 + kh + 2 * i);
    return t.v;
}

__device__ __forceinline__ v8f wmma16(v16h a, v16h b, v8f c) {
    return __builtin_amdgcn_wmma_f32_16x16x32_f16(false, a, false, b, (short)0, c,
                                                  false, false);
}

// ---------------------------------------------------------------- stage 1 ---
__global__ void cvt_x_f16(const float* __restrict__ x, _Float16* __restrict__ xh, int n) {
    int i = blockIdx.x * blockDim.x + threadIdx.x;
    if (i < n) xh[i] = (_Float16)x[i];
}

// whT[n][e] = W[e][n]  (n in [0,3072), e in [0,1024))
__global__ void cvt_w_t(const float* __restrict__ w, _Float16* __restrict__ whT) {
    int i = blockIdx.x * blockDim.x + threadIdx.x;
    if (i >= 3 * EMB * EMB) return;
    int ncol = i >> 10;          // /EMB
    int e    = i & (EMB - 1);
    whT[i] = (_Float16)w[(size_t)e * (3 * EMB) + ncol];
}

// additive key bias: log(w + tiny), overridden by +-9e15 for w==1 / w==0.
__global__ void bias_kernel(const float* __restrict__ wts, float* __restrict__ logw, int n) {
    int i = blockIdx.x * blockDim.x + threadIdx.x;
    if (i >= n) return;
    float w = wts[i];
    float v;
    if (w == 0.0f)      v = -9.0e15f;
    else if (w == 1.0f) v =  9.0e15f;
    else                v = __logf(w + 1.17549435e-38f);
    logw[i] = v;
}

// ---------------------------------------------------------------- stage 2 ---
// One wave computes a 32(M) x 64(N) tile of qkv = xh @ whT^T + b, then
// scatters into Q (scaled by 1/sqrt(hd)), K [bh][s][d], V^T [bh][d][s].
// Per 32-K chunk: 6 operand-tile loads (one clause group) feed 8 WMMAs.
__global__ void __launch_bounds__(256, 1)
qkv_gemm(const _Float16* __restrict__ xh,
         const _Float16* __restrict__ whT,
         const float*    __restrict__ bqkv,
         _Float16* __restrict__ Q,
         _Float16* __restrict__ Kk,
         _Float16* __restrict__ Vt) {
    const int wave = (blockIdx.x * blockDim.x + threadIdx.x) >> 5;
    const int lane = threadIdx.x & 31;
    const int NT   = (3 * EMB) / 64;                 // 48 column tiles
    const int rowTile = wave / NT;
    const int colTile = wave % NT;
    if (rowTile >= (BATCH * SEQ) / 32) return;       // wave-uniform guard
    const int m0 = rowTile * 32;
    const int n0 = colTile * 64;

    const _Float16* ar0 = xh + (size_t)m0 * EMB;
    const _Float16* ar1 = ar0 + (size_t)16 * EMB;
    const _Float16* br0 = whT + (size_t)(n0 +  0) * EMB;
    const _Float16* br1 = whT + (size_t)(n0 + 16) * EMB;
    const _Float16* br2 = whT + (size_t)(n0 + 32) * EMB;
    const _Float16* br3 = whT + (size_t)(n0 + 48) * EMB;

    v8f acc[2][4] = {};

    for (int k0 = 0; k0 < EMB; k0 += 32) {
        // issue all 6 tile loads up front (b128 clause); WMMAs consume them
        // in load order so the wait counters retire incrementally.
        const v16h a0 = load_tile(ar0 + k0, EMB);
        const v16h a1 = load_tile(ar1 + k0, EMB);
        const v16h b0 = load_tile(br0 + k0, EMB);
        const v16h b1 = load_tile(br1 + k0, EMB);
        const v16h b2 = load_tile(br2 + k0, EMB);
        const v16h b3 = load_tile(br3 + k0, EMB);
        acc[0][0] = wmma16(a0, b0, acc[0][0]);
        acc[1][0] = wmma16(a1, b0, acc[1][0]);
        acc[0][1] = wmma16(a0, b1, acc[0][1]);
        acc[1][1] = wmma16(a1, b1, acc[1][1]);
        acc[0][2] = wmma16(a0, b2, acc[0][2]);
        acc[1][2] = wmma16(a1, b2, acc[1][2]);
        acc[0][3] = wmma16(a0, b3, acc[0][3]);
        acc[1][3] = wmma16(a1, b3, acc[1][3]);
    }

    // epilogue: scatter into Q / K / V^T
    const int hi = lane >> 4, nl = lane & 15;
#pragma unroll
    for (int rs = 0; rs < 2; ++rs) {
#pragma unroll
        for (int t = 0; t < 4; ++t) {
            const int n    = n0 + t * 16 + nl;
            const int head = n / (3 * HD);           // /192
            const int rr   = n % (3 * HD);
            const float bn_ = bqkv[n];
#pragma unroll
            for (int r = 0; r < 8; ++r) {
                const int M  = m0 + rs * 16 + r + hi * 8;
                const int bb = M >> 11;              // /SEQ
                const int ss = M & (SEQ - 1);
                const float val = acc[rs][t][r] + bn_;
                const size_t bh = (size_t)(bb * HEADS + head);
                if (rr < HD)
                    Q[(bh * SEQ + ss) * HD + rr] = (_Float16)(val * 0.125f); // 1/sqrt(64)
                else if (rr < 2 * HD)
                    Kk[(bh * SEQ + ss) * HD + (rr - HD)] = (_Float16)val;
                else
                    Vt[(bh * HD + (rr - 2 * HD)) * SEQ + ss] = (_Float16)val;
            }
        }
    }
}

// ---------------------------------------------------------------- stage 3 ---
// Flash attention. One wave per (bh, 16-query tile). 32 keys per iteration:
//   QK^T (4 wmma) -> +bias -> online softmax -> P via LDS -> P@V (4 wmma).
// Next chunk's K/V cachelines are prefetched (global_prefetch_b8) while the
// softmax VALU work runs.
__global__ void __launch_bounds__(256, 1)
attn_kernel(const _Float16* __restrict__ Q,
            const _Float16* __restrict__ Kk,
            const _Float16* __restrict__ Vt,
            const float*    __restrict__ logw,
            float*          __restrict__ out) {
    __shared__ __align__(16) _Float16 pbuf[8][16 * 32];   // per-wave P tile
    const int wv   = threadIdx.x >> 5;
    const int lane = threadIdx.x & 31;
    const int wave = (blockIdx.x * blockDim.x + threadIdx.x) >> 5;
    const int bh   = wave >> 7;                       // / (SEQ/16)
    const int qt   = wave & 127;
    if (bh >= BATCH * HEADS) return;                  // wave-uniform guard
    const int b  = bh >> 4;
    const int h  = bh & (HEADS - 1);
    const int q0 = qt * 16;
    const int hi = lane >> 4, nl = lane & 15;

    const _Float16* qbase = Q + ((size_t)bh * SEQ + q0) * HD;
    const v16h aq0 = load_tile(qbase +  0, HD);       // K-dim d = 0..31
    const v16h aq1 = load_tile(qbase + 32, HD);       // K-dim d = 32..63

    float mrun[8], lrun[8];
    v8f o0 = {}, o1 = {}, o2 = {}, o3 = {};
#pragma unroll
    for (int r = 0; r < 8; ++r) { mrun[r] = -INFINITY; lrun[r] = 0.0f; }

    const float* lw = logw + (size_t)b * SEQ;

    for (int kb = 0; kb < SEQ; kb += 32) {
        // ---- S = Q K^T  (two 16x16 key sub-tiles) ----
        const _Float16* kb0 = Kk + ((size_t)bh * SEQ + kb) * HD;
        const _Float16* kb1 = kb0 + 16 * HD;
        const v16h bk00 = load_tile(kb0 +  0, HD);
        const v16h bk01 = load_tile(kb0 + 32, HD);
        const v16h bk10 = load_tile(kb1 +  0, HD);
        const v16h bk11 = load_tile(kb1 + 32, HD);
        v8f s0 = {}, s1 = {};
        s0 = wmma16(aq0, bk00, s0);
        s0 = wmma16(aq1, bk01, s0);
        s1 = wmma16(aq0, bk10, s1);
        s1 = wmma16(aq1, bk11, s1);

        // ---- prefetch next chunk's K/V rows into cache ----
        const int nkb = kb + 32;
        if (nkb < SEQ) {
            const _Float16* pk = Kk + ((size_t)bh * SEQ + nkb + nl) * HD;
            __builtin_prefetch(pk, 0, 3);                     // K rows nkb..+15
            __builtin_prefetch(pk + (size_t)16 * HD, 0, 3);   // K rows nkb+16..+31
            const _Float16* pv = Vt + ((size_t)bh * HD + nl) * SEQ + nkb;
            __builtin_prefetch(pv, 0, 3);
            __builtin_prefetch(pv + (size_t)16 * SEQ, 0, 3);
            __builtin_prefetch(pv + (size_t)32 * SEQ, 0, 3);
            __builtin_prefetch(pv + (size_t)48 * SEQ, 0, 3);
        }

        const float bias0 = lw[kb + nl];
        const float bias1 = lw[kb + 16 + nl];

        // ---- online softmax per row (rows live across 16-lane halves) ----
#pragma unroll
        for (int r = 0; r < 8; ++r) {
            float v0 = s0[r] + bias0;
            float v1 = s1[r] + bias1;
            float mx = fmaxf(v0, v1);
            mx = fmaxf(mx, __shfl_xor(mx, 1));
            mx = fmaxf(mx, __shfl_xor(mx, 2));
            mx = fmaxf(mx, __shfl_xor(mx, 4));
            mx = fmaxf(mx, __shfl_xor(mx, 8));
            const float mnew  = fmaxf(mrun[r], mx);
            const float scale = __expf(mrun[r] - mnew);   // first iter: exp(-inf)=0
            const float p0 = __expf(v0 - mnew);
            const float p1 = __expf(v1 - mnew);
            float sum = p0 + p1;
            sum += __shfl_xor(sum, 1);
            sum += __shfl_xor(sum, 2);
            sum += __shfl_xor(sum, 4);
            sum += __shfl_xor(sum, 8);
            lrun[r] = lrun[r] * scale + sum;
            mrun[r] = mnew;
            o0[r] *= scale; o1[r] *= scale; o2[r] *= scale; o3[r] *= scale;
            const int m = r + hi * 8;
            pbuf[wv][m * 32 + nl]      = (_Float16)p0;    // C-layout -> A-layout
            pbuf[wv][m * 32 + 16 + nl] = (_Float16)p1;    // via LDS transpose
        }
        __builtin_amdgcn_wave_barrier();                  // keep DS store->load order

        // ---- O += P V  (V^T stored [d][s] -> contiguous B-operand loads) ----
        const v16h ap = load_tile(&pbuf[wv][0], 32);
        const _Float16* vb = Vt + (size_t)bh * HD * SEQ + kb;
        o0 = wmma16(ap, load_tile(vb + (size_t) 0 * SEQ, SEQ), o0);
        o1 = wmma16(ap, load_tile(vb + (size_t)16 * SEQ, SEQ), o1);
        o2 = wmma16(ap, load_tile(vb + (size_t)32 * SEQ, SEQ), o2);
        o3 = wmma16(ap, load_tile(vb + (size_t)48 * SEQ, SEQ), o3);
    }

    // ---- normalize and write out[b][s][h*64+d] ----
    v8f o[4] = { o0, o1, o2, o3 };
#pragma unroll
    for (int r = 0; r < 8; ++r) {
        const int m  = r + hi * 8;
        const int ss = q0 + m;
        const float inv = 1.0f / lrun[r];
        float* op = out + ((size_t)b * SEQ + ss) * EMB + h * HD;
#pragma unroll
        for (int t = 0; t < 4; ++t)
            op[t * 16 + nl] = o[t][r] * inv;
    }
}

// ---------------------------------------------------------------- launch ----
extern "C" void kernel_launch(void* const* d_in, const int* in_sizes, int n_in,
                              void* d_out, int out_size, void* d_ws, size_t ws_size,
                              hipStream_t stream) {
    const float* x     = (const float*)d_in[0];   // (2, 2048, 1024)
    const float* wts   = (const float*)d_in[1];   // (2, 2048)
    const float* W_qkv = (const float*)d_in[2];   // (1024, 3072)
    const float* b_qkv = (const float*)d_in[3];   // (3072,)
    float* out = (float*)d_out;                   // (2, 2048, 1024)

    // workspace layout (f16 unless noted)
    char* ws = (char*)d_ws;
    const size_t nX  = (size_t)BATCH * SEQ * EMB;        // 4,194,304
    const size_t nW  = (size_t)3 * EMB * EMB;            // 3,145,728
    const size_t nPH = (size_t)BATCH * HEADS * SEQ * HD; // 4,194,304 per tensor
    _Float16* xh  = (_Float16*)ws;                 ws += nX  * 2;
    _Float16* whT = (_Float16*)ws;                 ws += nW  * 2;
    _Float16* Q   = (_Float16*)ws;                 ws += nPH * 2;
    _Float16* Kk  = (_Float16*)ws;                 ws += nPH * 2;
    _Float16* Vt  = (_Float16*)ws;                 ws += nPH * 2;
    float*    logw = (float*)ws;                   // BATCH*SEQ floats

    const int TB = 256;

    cvt_x_f16<<<(int)((nX + TB - 1) / TB), TB, 0, stream>>>(x, xh, (int)nX);
    cvt_w_t  <<<(int)((nW + TB - 1) / TB), TB, 0, stream>>>(W_qkv, whT);
    bias_kernel<<<(BATCH * SEQ + TB - 1) / TB, TB, 0, stream>>>(wts, logw, BATCH * SEQ);

    // 128 row tiles x 48 col tiles = 6144 waves; 8 waves/block
    qkv_gemm<<<6144 / 8, TB, 0, stream>>>(xh, whT, b_qkv, Q, Kk, Vt);

    // 32 bh x 128 query tiles = 4096 waves; 8 waves/block
    attn_kernel<<<4096 / 8, TB, 0, stream>>>(Q, Kk, Vt, logw, out);
}
